// MelSpecFeatures_3899830305080
// MI455X (gfx1250) — matrix operand order repaired and checked by reading, not compiled, not run on hardware
//
#include <hip/hip_runtime.h>
#include <math.h>

typedef _Float16 half16 __attribute__((ext_vector_type(16)));
typedef float    float8 __attribute__((ext_vector_type(8)));

#define NPOS 16384
#define NBATCH 16
#define FRAME 128
#define OUTW 81
#define POS_PER_BLOCK 128
#define THREADS 256

// Constant-table layout (all f16, pre-packed in WMMA lane order):
//  [0      .. 8191 ]  Bre tiles: t = kc*4+nc, 512 halves each (32 lanes x 16)
//  [8192   .. 16383]  Bim tiles: same order
//  [16384  .. 21503]  Wmel tiles: t2 = kc2*5+mc, 512 halves each
#define TABLE_HALFS 21504
#define BIM_OFF     8192
#define WMEL_OFF    16384

// 16-bit A-matrix (16x32) per-lane element -> K mapping (contiguous: K = e + 16*g, matches B tables)
__device__ __forceinline__ int kmapA(int e, int g) { return e + 16 * g; }

// Generate packed constant tables. Window is folded into the DFT matrices.
__device__ void fill_tables(_Float16* dst, int start, int stride) {
  const float PI2 = 6.28318530717958647692f;
  for (int h = start; h < TABLE_HALFS; h += stride) {
    float val;
    if (h < WMEL_OFF) {
      int isIm = h >= BIM_OFF;
      int hh = h & (BIM_OFF - 1);
      int t = hh >> 9; int lane = (hh >> 4) & 31; int e = hh & 15;
      int kc = t >> 2, nc = t & 3, g = lane >> 4;
      int K = kc * 32 + e + 16 * g;     // time sample 0..127 (B-layout K map)
      int F = nc * 16 + (lane & 15);    // 0..63 -> spectral bin F+1 (bin 0 has zero mel weight)
      float win = 0.5f - 0.5f * cosf(PI2 * (float)K / 128.0f);
      float ang = PI2 * (float)K * (float)(F + 1) / 128.0f;
      val = isIm ? (-win * sinf(ang)) : (win * cosf(ang));
    } else {
      int hh = h - WMEL_OFF;
      int t2 = hh >> 9; int lane = (hh >> 4) & 31; int e = hh & 15;
      int kc2 = t2 / 5, mc = t2 % 5, g = lane >> 4;
      int K2 = kc2 * 32 + e + 16 * g;   // spectral row 0..63 => bin K2+1
      int m  = mc * 16 + (lane & 15);   // mel column 0..79
      float spec_mel = 1127.0f * log1pf(125.0f * (float)(K2 + 1) / 700.0f);
      float mlo  = 1127.0f * log1pf(80.0f / 700.0f);
      float mhi  = 1127.0f * log1pf(7600.0f / 700.0f);
      float step = (mhi - mlo) / 81.0f;          // NUM_MEL + 2 = 82 edges
      float lower = mlo + (float)m * step;
      float ls = (spec_mel - lower) / step;
      float us = (lower + 2.0f * step - spec_mel) / step;
      val = fmaxf(0.0f, fminf(ls, us));
    }
    dst[h] = (_Float16)val;
  }
}

__global__ void init_tables_kernel(_Float16* tab) {
  int gid = blockIdx.x * blockDim.x + threadIdx.x;
  fill_tables(tab, gid, gridDim.x * blockDim.x);
}

template <bool USE_WS>
__global__ __launch_bounds__(THREADS)
void melspec_kernel(const float* __restrict__ x, float* __restrict__ out,
                    const _Float16* __restrict__ tab) {
  __shared__ __align__(32) _Float16 sB[TABLE_HALFS];        // 43008 B
  __shared__ float sx[POS_PER_BLOCK + FRAME];               // 1024 B (256 samples)
  __shared__ _Float16 sMag[8][16][66];                      // 16896 B, 66 stride vs bank conflicts

  const int tid = threadIdx.x;
  const int b   = blockIdx.y;
  const int p0  = blockIdx.x * POS_PER_BLOCK;

  if (USE_WS) {
    const uint4* src = (const uint4*)tab;
    uint4* dstv = (uint4*)sB;
    for (int i = tid; i < TABLE_HALFS / 8; i += THREADS) dstv[i] = src[i];
  } else {
    fill_tables(&sB[0], tid, THREADS);
  }
  {
    int gi = p0 + tid;                                      // zero-pad tail (rfft pad)
    sx[tid] = (gi < NPOS) ? x[(size_t)b * NPOS + gi] : 0.0f;
  }
  __syncthreads();

  const int wave = tid >> 5;
  const int lane = tid & 31;
  const int g    = lane >> 4;
  const int ln   = lane & 15;
  const int p    = p0 + wave * 16;                          // this wave's 16 frame positions

  // A tiles: overlapping frames (16 x 128) as f16, 4 K-chunks of 32
  half16 aA[4];
#pragma unroll
  for (int kc = 0; kc < 4; ++kc) {
#pragma unroll
    for (int e = 0; e < 16; ++e) {
      int K = kmapA(e, g) + kc * 32;
      aA[kc][e] = (_Float16)sx[wave * 16 + ln + K];
    }
  }

  // DFT (re/im) + magnitude, 4 bin-chunks of 16 (bins 1..64)
#pragma unroll
  for (int nc = 0; nc < 4; ++nc) {
    float8 accRe = {};
    float8 accIm = {};
#pragma unroll
    for (int kc = 0; kc < 4; ++kc) {
      const half16 bre = *(const half16*)&sB[(kc * 4 + nc) * 512 + lane * 16];
      const half16 bim = *(const half16*)&sB[BIM_OFF + (kc * 4 + nc) * 512 + lane * 16];
      accRe = __builtin_amdgcn_wmma_f32_16x16x32_f16(false, aA[kc], false, bre,
                                                     (short)0, accRe, false, false);
      accIm = __builtin_amdgcn_wmma_f32_16x16x32_f16(false, aA[kc], false, bim,
                                                     (short)0, accIm, false, false);
    }
#pragma unroll
    for (int r = 0; r < 8; ++r) {                           // D layout: lane->N, vgpr r -> M
      float mag = sqrtf(accRe[r] * accRe[r] + accIm[r] * accIm[r]);
      sMag[wave][g * 8 + r][nc * 16 + ln] = (_Float16)mag;
    }
  }
  asm volatile("" ::: "memory");  // same-wave LDS store->load ordering (DS is in-order per wave)

  // Transpose D-layout -> A-layout via LDS, then mel matmul mag(16x64) @ W(64x80)
  half16 a2[2];
#pragma unroll
  for (int kc2 = 0; kc2 < 2; ++kc2) {
#pragma unroll
    for (int e = 0; e < 16; ++e) {
      int K2 = kmapA(e, g) + kc2 * 32;
      a2[kc2][e] = sMag[wave][ln][K2];
    }
  }
#pragma unroll
  for (int mc = 0; mc < 5; ++mc) {
    float8 acc = {};
#pragma unroll
    for (int kc2 = 0; kc2 < 2; ++kc2) {
      const half16 bw = *(const half16*)&sB[WMEL_OFF + (kc2 * 5 + mc) * 512 + lane * 16];
      acc = __builtin_amdgcn_wmma_f32_16x16x32_f16(false, a2[kc2], false, bw,
                                                   (short)0, acc, false, false);
    }
#pragma unroll
    for (int r = 0; r < 8; ++r) {
      int m = g * 8 + r;
      out[((size_t)b * NPOS + (p + m)) * OUTW + 1 + mc * 16 + ln] = acc[r];
    }
  }

  // Passthrough column 0: out[b, p, 0] = x[b, p]
  if (lane < 16) {
    out[((size_t)b * NPOS + (p + lane)) * OUTW] = sx[wave * 16 + lane];
  }
}

extern "C" void kernel_launch(void* const* d_in, const int* in_sizes, int n_in,
                              void* d_out, int out_size, void* d_ws, size_t ws_size,
                              hipStream_t stream) {
  (void)in_sizes; (void)n_in; (void)out_size;
  const float* x = (const float*)d_in[0];
  float* out = (float*)d_out;
  const size_t tbytes = (size_t)TABLE_HALFS * 2u;
  dim3 grid(NPOS / POS_PER_BLOCK, NBATCH);
  if (d_ws != nullptr && ws_size >= tbytes) {
    _Float16* tab = (_Float16*)d_ws;
    init_tables_kernel<<<TABLE_HALFS / THREADS, THREADS, 0, stream>>>(tab);
    melspec_kernel<true><<<grid, THREADS, 0, stream>>>(x, out, tab);
  } else {
    melspec_kernel<false><<<grid, THREADS, 0, stream>>>(x, out, nullptr);
  }
}
